// RelativePositionalBias_7636451852511
// MI455X (gfx1250) — matrix operand order repaired and checked by reading, not compiled, not run on hardware
//
#include <hip/hip_runtime.h>
#include <hip/hip_bf16.h>
#include <stdint.h>

// B=4, N=1024, H=16, HID=64, P=2
#define BATCH 4
#define NTOK  1024
#define NHEAD 16
#define HID   64

typedef __attribute__((ext_vector_type(16))) _Float16 v16h;
typedef __attribute__((ext_vector_type(8)))  float    v8f;
typedef __attribute__((ext_vector_type(2)))  _Float16 v2h;
typedef __attribute__((ext_vector_type(2)))  float    v2f;

// One block per (b, i): 8 waves x 16-j tiles x 8 iterations cover j = 0..1023.
// GEMM1: h2 = relu(h1 @ W2 + b2)   -> 8x v_wmma_f32_16x16x32_f16 per tile
// GEMM2: bias^T = W3^T @ h2^T + b3 -> 2x v_wmma_f32_16x16x32_f16 per tile
__global__ __launch_bounds__(256) void relpos_bias_kernel(
    const float* __restrict__ pos,        // (B, N, 2)
    const uint8_t* __restrict__ kpm,      // (B, N) bool
    const float* __restrict__ W1,         // (2, 64)
    const float* __restrict__ b1,         // (64)
    const float* __restrict__ W2,         // (64, 64)
    const float* __restrict__ b2,         // (64)
    const float* __restrict__ W3,         // (64, 16)
    const float* __restrict__ b3,         // (16)
    float* __restrict__ out)              // (B, 16, N, N)
{
  const int i    = blockIdx.x;
  const int b    = blockIdx.y;
  const int tid  = threadIdx.x;
  const int wv   = tid >> 5;        // wave id (0..7)
  const int lane = tid & 31;
  const int ln   = lane & 15;
  const int hi   = lane >> 4;       // lane >= 16

  // Per-wave private transpose staging: h2 stored as [k][m] (64 x 16 f32).
  __shared__ __align__(64) float stage[8][HID * 16];
  float* sp_base = &stage[wv][0];

  // ---- Preload W1/b1 as A-layout half2 pairs --------------------------------
  // A (16x32 f16) lane layout: lane L holds row m=L%16; VGPR v holds K pair
  //   kp = 32c + 16*(v>=4) + 8*(L>=16) + 2*(v&3)
  v2h w1xp[2][8], w1yp[2][8], b1p[2][8];
  for (int c = 0; c < 2; ++c)
    for (int v = 0; v < 8; ++v) {
      const int kp = 32 * c + 16 * (v >> 2) + 8 * hi + 2 * (v & 3);
      w1xp[c][v] = (v2h){(_Float16)W1[kp],        (_Float16)W1[kp + 1]};
      w1yp[c][v] = (v2h){(_Float16)W1[HID + kp],  (_Float16)W1[HID + kp + 1]};
      b1p [c][v] = (v2h){(_Float16)b1[kp],        (_Float16)b1[kp + 1]};
    }

  // ---- Preload W2 in B-layout (32x16 f16 per chunk/ntile) -------------------
  // B lane layout: lane L holds col n=L%16; VGPR v holds K pair k = 32c + 16*(v>=4) + 8*hi + 2*(v&3)
  v16h w2b[2][4];
  for (int c = 0; c < 2; ++c)
    for (int nt = 0; nt < 4; ++nt) {
      v16h t = {};
      const int n = 16 * nt + ln;
      for (int v = 0; v < 8; ++v) {
        const int k = 32 * c + 16 * (v >> 2) + 8 * hi + 2 * (v & 3);
        t[2 * v]     = (_Float16)W2[k * HID + n];
        t[2 * v + 1] = (_Float16)W2[(k + 1) * HID + n];
      }
      w2b[c][nt] = t;
    }

  // ---- Preload W3^T in A-layout (rows = heads, K = hid) ---------------------
  v16h w3a[2];
  for (int c = 0; c < 2; ++c) {
    v16h t = {};
    for (int v = 0; v < 8; ++v) {
      const int kp = 32 * c + 16 * (v >> 2) + 8 * hi + 2 * (v & 3);
      t[2 * v]     = (_Float16)W3[kp * NHEAD + ln];
      t[2 * v + 1] = (_Float16)W3[(kp + 1) * NHEAD + ln];
    }
    w3a[c] = t;
  }

  // Bias broadcasts in the respective C/D layouts (hoisted out of the loop).
  v8f b2c[4];
  for (int nt = 0; nt < 4; ++nt) {
    const float bv = b2[16 * nt + ln];          // GEMM1 C: n = hid
    for (int r = 0; r < 8; ++r) b2c[nt][r] = bv;
  }
  v8f b3c;
  for (int r = 0; r < 8; ++r) b3c[r] = b3[r + 8 * hi];   // GEMM2 C: m = head

  const v2f pi  = *(const v2f*)(pos + ((size_t)b * NTOK + i) * 2);
  const bool mi = kpm[b * NTOK + i] != 0;

  for (int t = 0; t < 8; ++t) {
    const int j0 = wv * 128 + t * 16;
    const int j  = j0 + ln;                 // this lane's point (row m = ln)

    // Prefetch next tile's positions (global_prefetch_b8) to hide the
    // dependent load at the top of the next iteration.
    if (t < 7)
      __builtin_prefetch(pos + ((size_t)b * NTOK + j + 16) * 2, 0, 1);

    const v2f pj = *(const v2f*)(pos + ((size_t)b * NTOK + j) * 2);
    const _Float16 rxh = (_Float16)(pj.x - pi.x);
    const _Float16 ryh = (_Float16)(pj.y - pi.y);
    const v2h rx2 = {rxh, rxh}, ry2 = {ryh, ryh};

    // h1 = relu(rel @ W1 + b1) computed directly in A-register layout (f16).
    // Nested-FMA form: fma(w1x, rx, fma(w1y, ry, b1)) -> 2x v_pk_fma_f16.
    v16h a[2];
    for (int c = 0; c < 2; ++c) {
      v16h av = {};
      for (int v = 0; v < 8; ++v) {
        v2h h = w1xp[c][v] * rx2 + (w1yp[c][v] * ry2 + b1p[c][v]);
        _Float16 h0 = h[0] > (_Float16)0 ? h[0] : (_Float16)0;   // pk_max
        _Float16 h1v = h[1] > (_Float16)0 ? h[1] : (_Float16)0;
        av[2 * v] = h0;
        av[2 * v + 1] = h1v;
      }
      a[c] = av;
    }

    // GEMM1: h2 = relu(h1 @ W2 + b2); store transposed [k][m] into LDS.
    for (int nt = 0; nt < 4; ++nt) {
      v8f acc = b2c[nt];
      acc = __builtin_amdgcn_wmma_f32_16x16x32_f16(false, a[0], false,
                  w2b[0][nt], (short)0, acc, false, false);
      acc = __builtin_amdgcn_wmma_f32_16x16x32_f16(false, a[1], false,
                  w2b[1][nt], (short)0, acc, false, false);
      for (int r = 0; r < 8; ++r) acc[r] = __builtin_fmaxf(acc[r], 0.f);
      // D element (r, lane) = (m = r + 8*hi, k = 16*nt + ln).
      // [k][m] address: contiguous in r -> one aligned 32B store (2x ds b128).
      float* sp = sp_base + (16 * nt + ln) * 16 + 8 * hi;
      *(v8f*)sp = acc;
    }

    // Reload h2^T from LDS as the B operand (lane n = ln = point index).
    v16h b2m[2];
    for (int c = 0; c < 2; ++c) {
      v16h bb = {};
      for (int v = 0; v < 8; ++v) {
        const int k = 32 * c + 16 * (v >> 2) + 8 * hi + 2 * (v & 3);
        bb[2 * v]     = (_Float16)sp_base[k * 16 + ln];
        bb[2 * v + 1] = (_Float16)sp_base[(k + 1) * 16 + ln];
      }
      b2m[c] = bb;
    }

    // GEMM2 (transposed): D(m=head, n=j) = W3^T @ h2^T + b3.
    v8f d = __builtin_amdgcn_wmma_f32_16x16x32_f16(false, w3a[0], false,
                b2m[0], (short)0, b3c, false, false);
    d = __builtin_amdgcn_wmma_f32_16x16x32_f16(false, w3a[1], false,
                b2m[1], (short)0, d, false, false);

    // Mask + coalesced store: lanes 0..15 cover 16 consecutive j at head r,
    // lanes 16..31 at head r+8 -> two 64B segments per store instruction.
    const bool kill = mi || (kpm[b * NTOK + j] != 0);
    float* op = out + (((size_t)(b * NHEAD + 8 * hi)) * NTOK + i) * NTOK + j;
    for (int r = 0; r < 8; ++r) {
      const float val = kill ? 0.f : d[r];
      op[(size_t)r * NTOK * NTOK] = val;
    }
  }
}

extern "C" void kernel_launch(void* const* d_in, const int* in_sizes, int n_in,
                              void* d_out, int out_size, void* d_ws, size_t ws_size,
                              hipStream_t stream) {
  const float*   pos = (const float*)d_in[0];
  const uint8_t* kpm = (const uint8_t*)d_in[1];   // jax bool -> 1 byte/elem
  const float*   W1  = (const float*)d_in[2];
  const float*   b1  = (const float*)d_in[3];
  const float*   W2  = (const float*)d_in[4];
  const float*   b2  = (const float*)d_in[5];
  const float*   W3  = (const float*)d_in[6];
  const float*   b3  = (const float*)d_in[7];
  float* out = (float*)d_out;

  dim3 grid(NTOK, BATCH);   // one block per (i, b)
  dim3 block(256);          // 8 waves (wave32)
  relpos_bias_kernel<<<grid, block, 0, stream>>>(pos, kpm, W1, b1, W2, b2,
                                                 W3, b3, out);
}
